// GAT_TGNN_51453708206732
// MI455X (gfx1250) — matrix operand order repaired and checked by reading, not compiled, not run on hardware
//
#include <hip/hip_runtime.h>
#include <hip/hip_bf16.h>
#include <math.h>

typedef __bf16 bf16_t;
typedef __attribute__((ext_vector_type(16))) __bf16   v16bf;
typedef __attribute__((ext_vector_type(8)))  float    v8f;
typedef __attribute__((ext_vector_type(4)))  unsigned v4u;

union FragU { v4u q[2]; v16bf v; };

// ---------------------------------------------------------------------------
// Utility kernels
// ---------------------------------------------------------------------------
__global__ void k_fill_i32(int* __restrict__ p, int v, size_t n) {
    size_t i = (size_t)blockIdx.x * blockDim.x + threadIdx.x;
    if (i < n) p[i] = v;
}

__global__ void k_conv_bf16(const float* __restrict__ in, bf16_t* __restrict__ out, size_t n) {
    size_t i = (size_t)blockIdx.x * blockDim.x + threadIdx.x;
    if (i < n) out[i] = (bf16_t)in[i];
}

// in[R][C] -> out[C][R] (bf16), indexed by output element
__global__ void k_transpose_bf16(const float* __restrict__ in, bf16_t* __restrict__ out,
                                 int R, int C) {
    size_t i = (size_t)blockIdx.x * blockDim.x + threadIdx.x;
    size_t n = (size_t)R * C;
    if (i >= n) return;
    int c = (int)(i / R), r = (int)(i % R);
    out[i] = (bf16_t)in[(size_t)r * C + c];
}

// ---------------------------------------------------------------------------
// bf16 WMMA GEMM: C[M,N] = A[M,K] @ BT[N,K]^T   (BT is B transposed, row-major)
// One wave computes a 32x32 C tile as 2x2 WMMA 16x16x32 tiles.
// Fragment layout per CDNA5 ISA (16-bit A 16x32, B 32x16, f32 C 16x16).
// ---------------------------------------------------------------------------
__global__ __launch_bounds__(256)
void k_gemm_wmma_bf16(const bf16_t* __restrict__ A, const bf16_t* __restrict__ BT,
                      void* __restrict__ Cout, int M, int N, int K,
                      const float* __restrict__ bias, int relu, int out_bf16) {
    int tid   = blockIdx.x * blockDim.x + threadIdx.x;
    int wave  = tid >> 5;
    int lane  = tid & 31;
    int tilesN = N >> 5;
    int tilesM = M >> 5;
    if (wave >= tilesM * tilesN) return;
    int tm = wave / tilesN, tn = wave % tilesN;
    int m0 = tm << 5, n0 = tn << 5;
    int lhalf = lane >> 4;          // K-half selector
    int lrow  = lane & 15;          // row (A) / col (B,C)

    v8f acc00 = {}, acc01 = {}, acc10 = {}, acc11 = {};

    for (int k0 = 0; k0 < K; k0 += 32) {
        FragU a0, a1, b0, b1;
        // A frags: lanes 0-15 rows M=m0..m0+15 (K 0..7,16..23); lanes 16-31 K 8..15,24..31
        {
            const bf16_t* p = A + (size_t)(m0 + lrow) * K + k0 + lhalf * 8;
            a0.q[0] = *(const v4u*)p;
            a0.q[1] = *(const v4u*)(p + 16);
            const bf16_t* p1 = p + (size_t)16 * K;
            a1.q[0] = *(const v4u*)p1;
            a1.q[1] = *(const v4u*)(p1 + 16);
        }
        // B frags from BT: lane = N col; lanes 0-15 K 0..7,16..23; lanes 16-31 K 8..15,24..31
        {
            const bf16_t* p = BT + (size_t)(n0 + lrow) * K + k0 + lhalf * 8;
            b0.q[0] = *(const v4u*)p;
            b0.q[1] = *(const v4u*)(p + 16);
            const bf16_t* p1 = p + (size_t)16 * K;
            b1.q[0] = *(const v4u*)p1;
            b1.q[1] = *(const v4u*)(p1 + 16);
        }
        acc00 = __builtin_amdgcn_wmma_f32_16x16x32_bf16(false, a0.v, false, b0.v, (short)0, acc00, false, false);
        acc01 = __builtin_amdgcn_wmma_f32_16x16x32_bf16(false, a0.v, false, b1.v, (short)0, acc01, false, false);
        acc10 = __builtin_amdgcn_wmma_f32_16x16x32_bf16(false, a1.v, false, b0.v, (short)0, acc10, false, false);
        acc11 = __builtin_amdgcn_wmma_f32_16x16x32_bf16(false, a1.v, false, b1.v, (short)0, acc11, false, false);
    }

    // C layout: VGPR r -> lanes 0-15: (M=r, N=lane); lanes 16-31: (M=r+8, N=lane-16)
    auto store_tile = [&](v8f acc, int mb, int nb) {
        int col = nb + lrow;
        float bv = bias ? bias[col] : 0.0f;
#pragma unroll
        for (int r = 0; r < 8; ++r) {
            int row = mb + lhalf * 8 + r;
            float v = acc[r] + bv;
            if (relu) v = v > 0.0f ? v : 0.0f;
            size_t idx = (size_t)row * N + col;
            if (out_bf16) ((bf16_t*)Cout)[idx] = (bf16_t)v;
            else          ((float*)Cout)[idx]  = v;
        }
    };
    store_tile(acc00, m0,      n0);
    store_tile(acc01, m0,      n0 + 16);
    store_tile(acc10, m0 + 16, n0);
    store_tile(acc11, m0 + 16, n0 + 16);
}

// ---------------------------------------------------------------------------
// Per-(node,head) attention terms: al_src[n,h] = <h[n,h,:], a_src[h,:]> etc.
// One wave per (node, head), wave32 shuffle reduction.
// ---------------------------------------------------------------------------
template <typename T>
__global__ void k_node_terms(const T* __restrict__ h, const float* __restrict__ asrc,
                             const float* __restrict__ adst, float* __restrict__ als,
                             float* __restrict__ ald, int nodes, int heads, int hid) {
    int tid = blockIdx.x * blockDim.x + threadIdx.x;
    int wave = tid >> 5, lane = tid & 31;
    if (wave >= nodes * heads) return;
    int hd = wave % heads;
    const T*     hp = h + (size_t)wave * hid;
    const float* as = asrc + (size_t)hd * hid;
    const float* ad = adst + (size_t)hd * hid;
    float ss = 0.0f, dd = 0.0f;
    for (int c = lane; c < hid; c += 32) {
        float v = (float)hp[c];
        ss += v * as[c];
        dd += v * ad[c];
    }
#pragma unroll
    for (int o = 16; o > 0; o >>= 1) {
        ss += __shfl_xor(ss, o, 32);
        dd += __shfl_xor(dd, o, 32);
    }
    if (lane == 0) { als[wave] = ss; ald[wave] = dd; }
}

// ---------------------------------------------------------------------------
// CSR construction over the fixed edge set (E real edges + N self-loops).
// dst(e) = e < E ? ei[E+e] : e-E ;  src(e) = e < E ? ei[e] : e-E
// ---------------------------------------------------------------------------
__global__ void k_degree(const int* __restrict__ ei, int E_, int Nn, int* __restrict__ deg) {
    int e = blockIdx.x * blockDim.x + threadIdx.x;
    if (e >= E_ + Nn) return;
    int d = (e < E_) ? ei[E_ + e] : (e - E_);
    atomicAdd(&deg[d], 1);
}

// Single-block exclusive scan: rowptr[0..n] from deg[0..n); counter = copy of rowptr
__global__ __launch_bounds__(256)
void k_scan_excl(const int* __restrict__ deg, int* __restrict__ rowptr,
                 int* __restrict__ counter, int n) {
    __shared__ int part[257];
    int t = threadIdx.x;
    int chunk = (n + 255) / 256;
    int start = t * chunk;
    int end   = start + chunk; if (end > n) end = n;
    int sum = 0;
    for (int i = start; i < end; ++i) sum += deg[i];
    part[t] = sum;
    __syncthreads();
    if (t == 0) {
        int run = 0;
        for (int i = 0; i < 256; ++i) { int v = part[i]; part[i] = run; run += v; }
        part[256] = run;
    }
    __syncthreads();
    int base = part[t];
    for (int i = start; i < end; ++i) {
        rowptr[i] = base; counter[i] = base; base += deg[i];
    }
    if (start < n && end == n) rowptr[n] = base;
}

__global__ void k_bucket(const int* __restrict__ ei, int E_, int Nn,
                         int* __restrict__ counter, int* __restrict__ eidx) {
    int e = blockIdx.x * blockDim.x + threadIdx.x;
    if (e >= E_ + Nn) return;
    int d = (e < E_) ? ei[E_ + e] : (e - E_);
    int pos = atomicAdd(&counter[d], 1);
    eidx[pos] = e;
}

// Deterministic ordering: insertion-sort each dst bucket by edge id (avg degree ~9)
__global__ void k_sort_buckets(const int* __restrict__ rowptr, int* __restrict__ eidx, int n) {
    int d = blockIdx.x * blockDim.x + threadIdx.x;
    if (d >= n) return;
    int r0 = rowptr[d], r1 = rowptr[d + 1];
    for (int i = r0 + 1; i < r1; ++i) {
        int v = eidx[i], j = i - 1;
        while (j >= r0 && eidx[j] > v) { eidx[j + 1] = eidx[j]; --j; }
        eidx[j + 1] = v;
    }
}

// ---------------------------------------------------------------------------
// Fused segment-softmax + aggregation, gather-side, atomic-free.
// One wave per (dst, head). Pass 1: lane-parallel max over incident edges.
// Pass 2: serial edge loop, lanes own channels; acc[c] += exp(lg-m)*h[src,c];
// final scale by 1/sum. hid <= 512 (16 regs/lane).
// ---------------------------------------------------------------------------
template <typename T>
__global__ void k_gat_aggregate(const int* __restrict__ ei, int E_, int Nn,
                                const int* __restrict__ rowptr, const int* __restrict__ eidx,
                                const float* __restrict__ als, const float* __restrict__ ald,
                                const T* __restrict__ h, float* __restrict__ out,
                                int heads, int hid) {
    int tid = blockIdx.x * blockDim.x + threadIdx.x;
    int wave = tid >> 5, lane = tid & 31;
    if (wave >= Nn * heads) return;
    int d = wave / heads, hd = wave % heads;
    int r0 = rowptr[d], r1 = rowptr[d + 1];
    float aldv = ald[d * heads + hd];

    // pass 1: max logit (order-independent)
    float m = -3.0e38f;
    for (int j = r0 + lane; j < r1; j += 32) {
        int e = eidx[j];
        int s = (e < E_) ? ei[e] : (e - E_);
        float lg = als[s * heads + hd] + aldv;
        lg = lg > 0.0f ? lg : 0.2f * lg;
        m = fmaxf(m, lg);
    }
#pragma unroll
    for (int o = 16; o > 0; o >>= 1) m = fmaxf(m, __shfl_xor(m, o, 32));

    // pass 2: weighted accumulation, serial over sorted edges (deterministic)
    int nk = hid >> 5;              // 16 for hid=512
    float acc[16];
#pragma unroll
    for (int k = 0; k < 16; ++k) acc[k] = 0.0f;
    float ssum = 0.0f;
    for (int j = r0; j < r1; ++j) {
        int e = eidx[j];
        int s = (e < E_) ? ei[e] : (e - E_);
        float lg = als[s * heads + hd] + aldv;
        lg = lg > 0.0f ? lg : 0.2f * lg;
        float wgt = __expf(lg - m);
        ssum += wgt;
        const T* hp = h + ((size_t)s * heads + hd) * hid;
        for (int k = 0; k < nk; ++k)
            acc[k] += wgt * (float)hp[lane + 32 * k];
    }
    float inv = 1.0f / (ssum + 1e-16f);
    float* op = out + ((size_t)d * heads + hd) * hid;
    for (int k = 0; k < nk; ++k) op[lane + 32 * k] = acc[k] * inv;
}

// out_bf16[i] = elu(in[i] + bias[i % C])
__global__ void k_bias_elu_bf16(const float* __restrict__ in, const float* __restrict__ bias,
                                bf16_t* __restrict__ out, size_t n, int C) {
    size_t i = (size_t)blockIdx.x * blockDim.x + threadIdx.x;
    if (i >= n) return;
    float v = in[i] + bias[i % C];
    v = v > 0.0f ? v : expm1f(v);
    out[i] = (bf16_t)v;
}

// out_bf16[i] = in[i] + bias[i % C]
__global__ void k_bias_bf16(const float* __restrict__ in, const float* __restrict__ bias,
                            bf16_t* __restrict__ out, size_t n, int C) {
    size_t i = (size_t)blockIdx.x * blockDim.x + threadIdx.x;
    if (i >= n) return;
    out[i] = (bf16_t)(in[i] + bias[i % C]);
}

// Final: out[e,:7] = (h3[a]*h3[b]) @ fc2_W + fc2_b.  One wave per train edge.
__global__ void k_pair_fc2(const float* __restrict__ h3, const int* __restrict__ te, int Te,
                           const float* __restrict__ W, const float* __restrict__ b,
                           float* __restrict__ out, int hid) {
    int tid = blockIdx.x * blockDim.x + threadIdx.x;
    int wave = tid >> 5, lane = tid & 31;
    if (wave >= Te) return;
    int a  = te[wave];
    int c2 = te[Te + wave];
    const float* pa = h3 + (size_t)a * hid;
    const float* pb = h3 + (size_t)c2 * hid;
    float acc[7] = {0, 0, 0, 0, 0, 0, 0};
    for (int c = lane; c < hid; c += 32) {
        float p = pa[c] * pb[c];
        const float* w = W + (size_t)c * 7;
#pragma unroll
        for (int j = 0; j < 7; ++j) acc[j] += p * w[j];
    }
#pragma unroll
    for (int j = 0; j < 7; ++j)
        for (int o = 16; o > 0; o >>= 1) acc[j] += __shfl_xor(acc[j], o, 32);
    if (lane == 0) {
        float* op = out + (size_t)wave * 7;
#pragma unroll
        for (int j = 0; j < 7; ++j) op[j] = acc[j] + b[j];
    }
}

// ---------------------------------------------------------------------------
extern "C" void kernel_launch(void* const* d_in, const int* in_sizes, int n_in,
                              void* d_out, int out_size, void* d_ws, size_t ws_size,
                              hipStream_t stream) {
    const float* x   = (const float*)d_in[0];
    const int*   ei  = (const int*)d_in[1];
    const int*   te  = (const int*)d_in[2];
    const float* W1  = (const float*)d_in[3];
    const float* a1s = (const float*)d_in[4];
    const float* a1d = (const float*)d_in[5];
    const float* b1  = (const float*)d_in[6];
    const float* W2  = (const float*)d_in[7];
    const float* a2s = (const float*)d_in[8];
    const float* a2d = (const float*)d_in[9];
    const float* b2  = (const float*)d_in[10];
    const float* l1W = (const float*)d_in[11];
    const float* l1b = (const float*)d_in[12];
    const float* f2W = (const float*)d_in[13];
    const float* f2b = (const float*)d_in[14];

    const int IN = 128, HID = 512, H = 8;
    const int Nn = in_sizes[0] / IN;     // 8192
    const int Ee = in_sizes[1] / 2;      // 65536
    const int Te = in_sizes[2] / 2;      // 100000
    const int F1 = H * HID;              // 4096
    const int Ep = Ee + Nn;              // edges incl. self-loops

    // ---- workspace layout (out1 fp32 region reused for layer2/lin1 bufs) ----
    char* w = (char*)d_ws;
    auto ALN = [](size_t v) { return (v + 255) & ~(size_t)255; };
    size_t o = 0;
    bf16_t* h1b = (bf16_t*)(w + o);  o = ALN(o + (size_t)Nn * F1 * 2);   // 64 MB; later holds elu(h1) bf16
    size_t obig = o;                                                     // 128 MB region
    float*  out1 = (float*)(w + obig);
    float*  t2   = (float*)(w + obig);                                   // reuse after out1 dead
    float*  out2 = (float*)(w + obig + (size_t)Nn * HID * 4);
    bf16_t* h2b  = (bf16_t*)(w + obig + (size_t)Nn * HID * 8);
    float*  h3   = (float*)(w + obig + (size_t)Nn * HID * 10);
    o = ALN(obig + (size_t)Nn * F1 * 4);
    bf16_t* xb    = (bf16_t*)(w + o); o = ALN(o + (size_t)Nn * IN * 2);
    bf16_t* W1bT  = (bf16_t*)(w + o); o = ALN(o + (size_t)IN * F1 * 2);
    bf16_t* W2bT  = (bf16_t*)(w + o); o = ALN(o + (size_t)F1 * HID * 2);
    bf16_t* l1WbT = (bf16_t*)(w + o); o = ALN(o + (size_t)HID * HID * 2);
    float* al1s = (float*)(w + o); o = ALN(o + (size_t)Nn * H * 4);
    float* al1d = (float*)(w + o); o = ALN(o + (size_t)Nn * H * 4);
    float* al2s = (float*)(w + o); o = ALN(o + (size_t)Nn * 4);
    float* al2d = (float*)(w + o); o = ALN(o + (size_t)Nn * 4);
    int* deg     = (int*)(w + o); o = ALN(o + (size_t)Nn * 4);
    int* rowptr  = (int*)(w + o); o = ALN(o + (size_t)(Nn + 1) * 4);
    int* counter = (int*)(w + o); o = ALN(o + (size_t)Nn * 4);
    int* eidx    = (int*)(w + o); o = ALN(o + (size_t)Ep * 4);

    const int B = 256;
    auto blk = [&](size_t threads) { return dim3((unsigned)((threads + B - 1) / B)); };

    // ---- CSR build (graph is shared by both GAT layers) ----
    k_fill_i32<<<blk((size_t)Nn), B, 0, stream>>>(deg, 0, (size_t)Nn);
    k_degree<<<blk((size_t)Ep), B, 0, stream>>>(ei, Ee, Nn, deg);
    k_scan_excl<<<1, 256, 0, stream>>>(deg, rowptr, counter, Nn);
    k_bucket<<<blk((size_t)Ep), B, 0, stream>>>(ei, Ee, Nn, counter, eidx);
    k_sort_buckets<<<blk((size_t)Nn), B, 0, stream>>>(rowptr, eidx, Nn);

    // ---- precision conversion / weight transposes (bf16 for WMMA) ----
    k_conv_bf16<<<blk((size_t)Nn * IN), B, 0, stream>>>(x, xb, (size_t)Nn * IN);
    k_transpose_bf16<<<blk((size_t)IN * F1),  B, 0, stream>>>(W1,  W1bT,  IN,  F1);
    k_transpose_bf16<<<blk((size_t)F1 * HID), B, 0, stream>>>(W2,  W2bT,  F1,  HID);
    k_transpose_bf16<<<blk((size_t)HID * HID),B, 0, stream>>>(l1W, l1WbT, HID, HID);

    // ---- layer 1: h1 = x @ W1 (bf16 out; 64 MB -> lives in L2 for the gathers) ----
    size_t waves1 = (size_t)(Nn / 32) * (F1 / 32);
    k_gemm_wmma_bf16<<<blk(waves1 * 32), B, 0, stream>>>(xb, W1bT, h1b, Nn, F1, IN,
                                                         nullptr, 0, /*out_bf16=*/1);
    k_node_terms<bf16_t><<<blk((size_t)Nn * H * 32), B, 0, stream>>>(h1b, a1s, a1d,
                                                                     al1s, al1d, Nn, H, HID);
    k_gat_aggregate<bf16_t><<<blk((size_t)Nn * H * 32), B, 0, stream>>>(
        ei, Ee, Nn, rowptr, eidx, al1s, al1d, h1b, out1, H, HID);
    k_bias_elu_bf16<<<blk((size_t)Nn * F1), B, 0, stream>>>(out1, b1, h1b, (size_t)Nn * F1, F1);

    // ---- layer 2: t2 = elu(h1) @ W2 (fp32 out), heads=1 ----
    size_t waves2 = (size_t)(Nn / 32) * (HID / 32);
    k_gemm_wmma_bf16<<<blk(waves2 * 32), B, 0, stream>>>(h1b, W2bT, t2, Nn, HID, F1,
                                                         nullptr, 0, /*out_bf16=*/0);
    k_node_terms<float><<<blk((size_t)Nn * 32), B, 0, stream>>>(t2, a2s, a2d,
                                                                al2s, al2d, Nn, 1, HID);
    k_gat_aggregate<float><<<blk((size_t)Nn * 32), B, 0, stream>>>(
        ei, Ee, Nn, rowptr, eidx, al2s, al2d, t2, out2, 1, HID);
    k_bias_bf16<<<blk((size_t)Nn * HID), B, 0, stream>>>(out2, b2, h2b, (size_t)Nn * HID, HID);

    // ---- lin1 (fused bias + relu) ----
    size_t waves3 = (size_t)(Nn / 32) * (HID / 32);
    k_gemm_wmma_bf16<<<blk(waves3 * 32), B, 0, stream>>>(h2b, l1WbT, h3, Nn, HID, HID,
                                                         l1b, /*relu=*/1, /*out_bf16=*/0);

    // ---- pair product + fc2 ----
    k_pair_fc2<<<blk((size_t)Te * 32), B, 0, stream>>>(h3, te, Te, f2W, f2b,
                                                       (float*)d_out, HID);
}
